// MinimalPerformerAttention_58763742544009
// MI455X (gfx1250) — matrix operand order repaired and checked by reading, not compiled, not run on hardware
//
#include <hip/hip_runtime.h>
#include <hip/hip_bf16.h>

typedef __attribute__((ext_vector_type(16))) __bf16 v16bf;
typedef __attribute__((ext_vector_type(8)))  float  v8f;
typedef __attribute__((ext_vector_type(4)))  unsigned int u32x4;

#define B_    4
#define H_    16
#define S_    2048
#define DH_   64
#define NF_   64
#define DIM_  1024
#define NROWS 131072      // B_*H_*S_
#define NCH   2048        // total chunks = B_*H_*32
#define CHPB  32          // chunks per (b,h)

// ---------------------------------------------------------------------------
// fp32 -> bf16 convert
// ---------------------------------------------------------------------------
__global__ __launch_bounds__(256) void k_f32_to_bf16(const float* __restrict__ s,
                                                     __bf16* __restrict__ d, long n) {
  long i = (long)blockIdx.x * 256 + threadIdx.x;
  if (i < n) d[i] = (__bf16)s[i];
}

// ---------------------------------------------------------------------------
// C[M,N](f32) = A[M,K](bf16) @ B[N,K]^T(bf16)   via v_wmma_f32_16x16x32_bf16
// WG = 256 thr = 8 waves (4 along M x 2 along N). WG tile 256x64.
// Wave tile 64x32 = 4x2 WMMA tiles -> 8 WMMA per K-step vs 12 ds_load_b128
// (768B LDS traffic per WMMA). Requires M%256==0, N%64==0, K%32==0.
// ---------------------------------------------------------------------------
__global__ __launch_bounds__(256) void k_gemm_bf16(const __bf16* __restrict__ A,
                                                   const __bf16* __restrict__ Bm,
                                                   float* __restrict__ C,
                                                   int M, int N, int K) {
  __shared__ __bf16 As[256][40];   // +8 bf16 pad: row stride 80B (16B aligned)
  __shared__ __bf16 Bs[64][40];
  const int tid   = threadIdx.x;
  const int lane  = tid & 31;
  const int wv    = tid >> 5;
  const int waveM = wv & 3;            // 4 waves along M
  const int waveN = wv >> 2;           // 2 waves along N
  const int wgM   = blockIdx.x * 256;
  const int wgN   = blockIdx.y * 64;

  v8f acc[4][2] = {};

  const int brow  = tid >> 2;           // 0..63
  const int bq    = (tid & 3) * 8;      // 8 bf16 = 16B each

  // fragment gather coords (ISA 7.12.2 16-bit A 16x32 / B 32x16 layouts)
  const int lm = lane & 15;
  const int kb = (lane >> 4) * 8;       // A: K base 0/8 (then +16)
  const int kh = (lane >> 4) * 16;      // unused (B also uses kb)

  for (int k0 = 0; k0 < K; k0 += 32) {
    { // stage tiles to LDS: one A row (64B) per thread, one B quarter-row
      const __bf16* sa = A + (size_t)(wgM + tid) * K + k0;
      u32x4* da = (u32x4*)&As[tid][0];
      da[0] = ((const u32x4*)sa)[0];
      da[1] = ((const u32x4*)sa)[1];
      da[2] = ((const u32x4*)sa)[2];
      da[3] = ((const u32x4*)sa)[3];
      const __bf16* sb = Bm + (size_t)(wgN + brow) * K + k0 + bq;
      *(u32x4*)&Bs[brow][bq] = *(const u32x4*)sb;
    }
    __syncthreads();

    v16bf af[4], bfr[2];
#pragma unroll
    for (int mt = 0; mt < 4; ++mt) {
      const __bf16* p = &As[waveM * 64 + mt * 16 + lm][0];
      u32x4* d = (u32x4*)&af[mt];
      d[0] = *(const u32x4*)(p + kb);          // K = kb..kb+7
      d[1] = *(const u32x4*)(p + 16 + kb);     // K = 16+kb..16+kb+7
    }
#pragma unroll
    for (int nt = 0; nt < 2; ++nt) {
      const __bf16* p = &Bs[waveN * 32 + nt * 16 + lm][0];
      u32x4* d = (u32x4*)&bfr[nt];
      d[0] = *(const u32x4*)(p + kb);          // K = kb..kb+7
      d[1] = *(const u32x4*)(p + 16 + kb);     // K = 16+kb..16+kb+7
    }
#pragma unroll
    for (int mt = 0; mt < 4; ++mt)
#pragma unroll
      for (int nt = 0; nt < 2; ++nt)
        acc[mt][nt] = __builtin_amdgcn_wmma_f32_16x16x32_bf16(
            false, af[mt], false, bfr[nt], (short)0, acc[mt][nt], false, false);
    __syncthreads();
  }

  const int hi8 = (lane >> 4) * 8;   // C layout: m = r + 8*(lane>>4), n = lane&15
#pragma unroll
  for (int mt = 0; mt < 4; ++mt)
#pragma unroll
    for (int nt = 0; nt < 2; ++nt) {
      const int cm = wgM + waveM * 64 + mt * 16 + hi8;
      const int cn = wgN + waveN * 32 + nt * 16 + (lane & 15);
      float* cp = C + (size_t)cm * N + cn;
#pragma unroll
      for (int r = 0; r < 8; ++r) cp[(size_t)r * N] = acc[mt][nt][r];
    }
}

// ---------------------------------------------------------------------------
// Gather (bug-faithful reshape) + scale + diag for q/k.  1 thread per row.
// row r = (b,h,s): src = lin[b, h*128 + s/16, (s%16)*64 + d]
// ---------------------------------------------------------------------------
__global__ __launch_bounds__(256) void k_gather_qk(const float* __restrict__ lin,
                                                   __bf16* __restrict__ g,
                                                   float* __restrict__ diag) {
  const int r = blockIdx.x * 256 + threadIdx.x;
  if (r >= NROWS) return;
  const int bi = r >> 15, rem = r & 32767, hi = rem >> 11, si = rem & 2047;
  const float* src = lin + ((size_t)bi * S_ + hi * 128 + (si >> 4)) * DIM_ + (si & 15) * 64;
  __bf16* dst = g + (size_t)r * 64;
  const float dn = 0.35355339059327373f;  // 64^-0.25
  float ss = 0.f;
#pragma unroll 8
  for (int d = 0; d < 64; ++d) {
    const float v = dn * src[d];
    ss += v * v;
    dst[d] = (__bf16)v;
  }
  diag[r] = ss * 0.0625f;   // * dn^2 / 2 = * 0.125/2
}

__global__ __launch_bounds__(256) void k_gather_v(const float* __restrict__ lin,
                                                  float* __restrict__ vg) {
  const int idx = blockIdx.x * 256 + threadIdx.x;   // < 8388608
  const int r = idx >> 6, d = idx & 63;
  const int bi = r >> 15, rem = r & 32767, hi = rem >> 11, si = rem & 2047;
  vg[idx] = lin[((size_t)bi * S_ + hi * 128 + (si >> 4)) * DIM_ + (si & 15) * 64 + d];
}

// ---------------------------------------------------------------------------
// Feature map: out = 0.125 * (exp(dash - [rowmax] - diag) + 1e-4)
// ---------------------------------------------------------------------------
__global__ __launch_bounds__(256) void k_feature(const float* __restrict__ dash,
                                                 const float* __restrict__ diag,
                                                 float* __restrict__ out, int is_query) {
  const int r = blockIdx.x * 256 + threadIdx.x;
  if (r >= NROWS) return;
  const float* ds = dash + (size_t)r * 64;
  float* o = out + (size_t)r * 64;
  const float dg = diag[r];
  float m = 0.f;
  if (is_query) {
    m = -1e30f;
    for (int j = 0; j < 64; ++j) m = fmaxf(m, ds[j]);
  }
  for (int j = 0; j < 64; ++j)
    o[j] = 0.125f * (__expf(ds[j] - m - dg) + 1e-4f);
}

// ---------------------------------------------------------------------------
// Chunked linear attention. Chunk = 64 steps; rows of chunk c are c*64..c*64+63
// Pass 1: per-chunk KV = K^T V (64x64) and Ksum (64)
// ---------------------------------------------------------------------------
__global__ __launch_bounds__(256) void k_chunk_sums(const float* __restrict__ kd,
                                                    const float* __restrict__ vg,
                                                    float* __restrict__ KVc,
                                                    float* __restrict__ Ksum) {
  __shared__ float Kl[64 * 64], Vl[64 * 64];
  const int c = blockIdx.x, tid = threadIdx.x;
  const size_t base = (size_t)c * 4096;
  for (int i = tid; i < 4096; i += 256) { Kl[i] = kd[base + i]; Vl[i] = vg[base + i]; }
  __syncthreads();
  const int f = tid >> 2, d0 = (tid & 3) * 16;
  float a[16] = {};
  float ks = 0.f;
  for (int t = 0; t < 64; ++t) {
    const float kf = Kl[t * 64 + f];
    ks += kf;
#pragma unroll
    for (int dd = 0; dd < 16; ++dd) a[dd] += kf * Vl[t * 64 + d0 + dd];
  }
  float* o = KVc + base + f * 64 + d0;
#pragma unroll
  for (int dd = 0; dd < 16; ++dd) o[dd] = a[dd];
  if ((tid & 3) == 0) Ksum[(size_t)c * 64 + f] = ks;
}

// Pass 2: exclusive prefix over the 32 chunks of each (b,h). 1 WG per (b,h).
__global__ __launch_bounds__(256) void k_chunk_prefix(const float* __restrict__ KVc,
                                                      const float* __restrict__ Ksum,
                                                      float* __restrict__ KVp,
                                                      float* __restrict__ Kcum) {
  const int bh = blockIdx.x, tid = threadIdx.x;
  const int cell0 = tid * 16;
  float a[16] = {};
  float ka = 0.f;
  for (int c = 0; c < CHPB; ++c) {
    const size_t base = ((size_t)bh * CHPB + c) * 4096;
#pragma unroll
    for (int i = 0; i < 16; ++i) KVp[base + cell0 + i] = a[i];
#pragma unroll
    for (int i = 0; i < 16; ++i) a[i] += KVc[base + cell0 + i];
    if (tid < 64) {
      const size_t kb = ((size_t)bh * CHPB + c) * 64 + tid;
      Kcum[kb] = ka;
      ka += Ksum[kb];
    }
  }
}

// Pass 3: per chunk: A = tril(QK^T); denom = Q.kcum + rowsum(A);
//         out = (Q@KVprev + A@V) / max(denom, 1e-6)
__global__ __launch_bounds__(256) void k_chunk_out(const float* __restrict__ qd,
                                                   const float* __restrict__ kd,
                                                   const float* __restrict__ vg,
                                                   const float* __restrict__ KVp,
                                                   const float* __restrict__ Kcum,
                                                   float* __restrict__ so) {
  __shared__ float Q[4096], Kl[4096], V[4096], KVs[4096];
  __shared__ float Am[64 * 65];
  __shared__ float kc[64], den[64];
  const int c = blockIdx.x, tid = threadIdx.x;
  const size_t base = (size_t)c * 4096;
  for (int i = tid; i < 4096; i += 256) {
    Q[i] = qd[base + i]; Kl[i] = kd[base + i];
    V[i] = vg[base + i]; KVs[i] = KVp[base + i];
  }
  if (tid < 64) kc[tid] = Kcum[(size_t)c * 64 + tid];
  __syncthreads();

  const int i = tid >> 2, g0 = (tid & 3) * 16;
  for (int j = g0; j < g0 + 16; ++j) {
    float s = 0.f;
    if (j <= i) {
      for (int d = 0; d < 64; ++d) s += Q[i * 64 + d] * Kl[j * 64 + d];
    }
    Am[i * 65 + j] = s;
  }
  __syncthreads();

  if (tid < 64) {
    float s = 0.f;
    for (int f = 0; f < 64; ++f) s += Q[tid * 64 + f] * kc[f];
    for (int j = 0; j <= tid; ++j) s += Am[tid * 65 + j];
    den[tid] = fmaxf(s, 1e-6f);
  }
  __syncthreads();

  float acc[16] = {};
  for (int f = 0; f < 64; ++f) {
    const float qf = Q[i * 64 + f];
#pragma unroll
    for (int dd = 0; dd < 16; ++dd) acc[dd] += qf * KVs[f * 64 + g0 + dd];
  }
  for (int j = 0; j <= i; ++j) {
    const float aij = Am[i * 65 + j];
#pragma unroll
    for (int dd = 0; dd < 16; ++dd) acc[dd] += aij * V[j * 64 + g0 + dd];
  }
  const float inv = 1.0f / den[i];
#pragma unroll
  for (int dd = 0; dd < 16; ++dd) so[base + i * 64 + g0 + dd] = acc[dd] * inv;
}

// ---------------------------------------------------------------------------
// (b,h,s,dh) -> (b,s,h*dh) transpose + bf16 for the final GEMM
// ---------------------------------------------------------------------------
__global__ __launch_bounds__(256) void k_unpermute(const float* __restrict__ post,
                                                   __bf16* __restrict__ preb) {
  const int o = blockIdx.x * 256 + threadIdx.x;   // < 8388608
  const int di = o & 63, hi = (o >> 6) & 15, si = (o >> 10) & 2047, bi = o >> 21;
  preb[o] = (__bf16)post[(((size_t)(bi * 16 + hi) * S_) + si) * 64 + di];
}

// ---------------------------------------------------------------------------
extern "C" void kernel_launch(void* const* d_in, const int* in_sizes, int n_in,
                              void* d_out, int out_size, void* d_ws, size_t ws_size,
                              hipStream_t stream) {
  (void)in_sizes; (void)n_in; (void)out_size; (void)ws_size;
  const float* x     = (const float*)d_in[0];
  const float* Wq    = (const float*)d_in[1];
  const float* Wk    = (const float*)d_in[2];
  const float* Wv    = (const float*)d_in[3];
  const float* proj  = (const float*)d_in[4];
  const float* Wpost = (const float*)d_in[5];
  const float* Wout  = (const float*)d_in[6];

  char* w = (char*)d_ws;
  size_t o = 0;
  auto take = [&](size_t bytes) -> char* {
    char* p = w + o;
    o += (bytes + 255) & ~(size_t)255;
    return p;
  };

  __bf16* xb     = (__bf16*)take(16777216);
  __bf16* Wqb    = (__bf16*)take(2097152);
  __bf16* Wkb    = (__bf16*)take(2097152);
  __bf16* Wvb    = (__bf16*)take(2097152);
  __bf16* Woutb  = (__bf16*)take(2097152);
  __bf16* projb  = (__bf16*)take(8192);
  __bf16* Wpostb = (__bf16*)take(8192);
  float*  diagq  = (float*)take(524288);
  float*  diagk  = (float*)take(524288);
  float*  vg     = (float*)take(33554432);
  char*   R1     = take(100663296);   // lin -> dash -> chunk-states -> preb
  char*   R2     = take(33554432);    // qg+kg -> scan out
  char*   R3     = take(67108864);    // qdash+kdash -> sob+post

  float*  q_lin = (float*)R1;
  float*  k_lin = (float*)(R1 + 33554432);
  float*  v_lin = (float*)(R1 + 67108864);
  float*  dashq = (float*)R1;                       // reuse after gathers
  float*  dashk = (float*)(R1 + 33554432);
  float*  KVc   = (float*)R1;                       // reuse after features
  float*  KVp   = (float*)(R1 + 33554432);
  float*  Ksumc = (float*)(R1 + 67108864);
  float*  Kcum  = (float*)(R1 + 67108864 + 524288);
  __bf16* preb  = (__bf16*)R1;                      // reuse after chunk pass

  __bf16* qg = (__bf16*)R2;
  __bf16* kg = (__bf16*)(R2 + 16777216);
  float*  so = (float*)R2;                          // reuse after dash GEMMs

  float*  qdash = (float*)R3;
  float*  kdash = (float*)(R3 + 33554432);
  __bf16* sob   = (__bf16*)R3;                      // reuse after chunk pass
  float*  post  = (float*)(R3 + 16777216);

  auto cvt = [&](const float* s, __bf16* d, long n) {
    k_f32_to_bf16<<<dim3((unsigned)((n + 255) / 256)), dim3(256), 0, stream>>>(s, d, n);
  };
  auto gemm = [&](const __bf16* A, const __bf16* Bm, float* C, int M, int N, int K) {
    k_gemm_bf16<<<dim3(M / 256, N / 64), dim3(256), 0, stream>>>(A, Bm, C, M, N, K);
  };

  // bf16 casts of x and weights
  cvt(x, xb, 8388608);
  cvt(Wq, Wqb, 1048576);
  cvt(Wk, Wkb, 1048576);
  cvt(Wv, Wvb, 1048576);
  cvt(Wout, Woutb, 1048576);
  cvt(proj, projb, 4096);
  cvt(Wpost, Wpostb, 4096);

  // q/k/v projections: (8192,1024) = xb @ W^T
  gemm(xb, Wqb, q_lin, 8192, 1024, 1024);
  gemm(xb, Wkb, k_lin, 8192, 1024, 1024);
  gemm(xb, Wvb, v_lin, 8192, 1024, 1024);

  // bug-faithful reshape gather + data_normalizer + diag
  k_gather_qk<<<dim3(512), dim3(256), 0, stream>>>(q_lin, qg, diagq);
  k_gather_qk<<<dim3(512), dim3(256), 0, stream>>>(k_lin, kg, diagk);
  k_gather_v<<<dim3(32768), dim3(256), 0, stream>>>(v_lin, vg);

  // random-feature projection: (131072,64) = g @ proj^T
  gemm(qg, projb, dashq, NROWS, 64, 64);
  gemm(kg, projb, dashk, NROWS, 64, 64);

  // softmax kernel feature map
  k_feature<<<dim3(512), dim3(256), 0, stream>>>(dashq, diagq, qdash, 1);
  k_feature<<<dim3(512), dim3(256), 0, stream>>>(dashk, diagk, kdash, 0);

  // chunked causal linear attention
  k_chunk_sums<<<dim3(NCH), dim3(256), 0, stream>>>(kdash, vg, KVc, Ksumc);
  k_chunk_prefix<<<dim3(B_ * H_), dim3(256), 0, stream>>>(KVc, Ksumc, KVp, Kcum);
  k_chunk_out<<<dim3(NCH), dim3(256), 0, stream>>>(qdash, kdash, vg, KVp, Kcum, so);

  // post projection: (131072,64) = so @ Wpost^T
  cvt(so, sob, 8388608);
  gemm(sob, Wpostb, post, NROWS, 64, 64);

  // (b,h,s,dh) -> (b,s,h*dh), then final: (8192,1024) = pre @ Wout^T
  k_unpermute<<<dim3(32768), dim3(256), 0, stream>>>(post, preb);
  gemm(preb, Woutb, (float*)d_out, 8192, 1024, 1024);
}
